// CoarseTransformer_64252710748528
// MI455X (gfx1250) — matrix operand order repaired and checked by reading, not compiled, not run on hardware
//
#include <hip/hip_runtime.h>
#include <hip/hip_bf16.h>

typedef __bf16 bf16_t;
typedef __attribute__((ext_vector_type(16))) __bf16 v16bf;
typedef __attribute__((ext_vector_type(8)))  __bf16 v8bf;
typedef __attribute__((ext_vector_type(8)))  float  v8f;

#define NTOK 1025      // 1 + 256 + 768
#define ROWS 2050      // B * NTOK
#define DIMM 1024
#define FFI  2730
#define FFIK 2752      // FFI padded up to multiple of 32 (K-dim padding, zero-filled)
#define W1N  5460      // 2*FFI
#define W1NP 5504      // padded to multiple of 128 (zero-filled)
#define SEMV 501
#define SEMP 640       // padded to multiple of 128 (zero-filled)
#define CBV  1025
#define CBP  1152      // padded to multiple of 128 (zero-filled)

__device__ __forceinline__ v8f wmma_bf16(v16bf a, v16bf b, v8f c) {
    return __builtin_amdgcn_wmma_f32_16x16x32_bf16(false, a, false, b, (short)0, c, false, false);
}

// ---------------- embedding gather ----------------
__global__ __launch_bounds__(256) void k_embed(
    const int* __restrict__ sem_ids, const int* __restrict__ coarse_ids,
    const float* __restrict__ start_tok, const float* __restrict__ sem_emb,
    const float* __restrict__ coarse_emb, float* __restrict__ x)
{
    int row = blockIdx.x;                 // 0..2049
    int b = row / NTOK, t = row % NTOK;
    const float* src;
    if (t == 0) {
        src = start_tok;
    } else if (t <= 256) {
        src = sem_emb + (long)sem_ids[b * 256 + (t - 1)] * DIMM;
    } else {
        int j = t - 257;
        int cid = coarse_ids[b * 768 + j] + (j % 3) * 1024;
        src = coarse_emb + (long)cid * DIMM;
    }
    float* dst = x + (long)row * DIMM;
    for (int d = threadIdx.x; d < DIMM; d += 256) dst[d] = src[d];
}

// ---------------- relative position bias table: bt[h][d], d = max(i-j,0) ----
__global__ __launch_bounds__(256) void k_biastab(
    const float* __restrict__ rel_emb, float* __restrict__ btab)
{
    int tid = blockIdx.x * 256 + threadIdx.x;
    if (tid >= 8 * NTOK) return;
    int h = tid / NTOK, d = tid % NTOK;
    int bucket;
    if (d < 16) {
        bucket = d;
    } else {
        float val = logf((float)d * (1.0f / 16.0f)) / logf(8.0f) * 16.0f;
        int vl = 16 + (int)val;
        bucket = vl < 31 ? vl : 31;
    }
    btab[tid] = rel_emb[bucket * 8 + h];
}

// ---------------- f32 -> bf16 convert (contiguous, no padding) ----------------
__global__ __launch_bounds__(256) void k_cvt(
    const float* __restrict__ src, bf16_t* __restrict__ dst, long n)
{
    long i0 = ((long)blockIdx.x * 256 + threadIdx.x) * 4;
    #pragma unroll
    for (int u = 0; u < 4; u++) {
        long i = i0 + u;
        if (i < n) dst[i] = (bf16_t)src[i];
    }
}

// f32 (Ks x Ns) -> bf16 (Kp x Np), zero-filled padding (dst-indexed)
__global__ __launch_bounds__(256) void k_cvt_pad2d(
    const float* __restrict__ src, bf16_t* __restrict__ dst,
    int Ks, int Ns, int Kp, int Np)
{
    long o = (long)blockIdx.x * 256 + threadIdx.x;
    long total = (long)Kp * Np;
    if (o >= total) return;
    long r = o / Np, c = o % Np;
    float v = (r < Ks && c < Ns) ? src[r * Ns + c] : 0.f;
    dst[o] = (bf16_t)v;
}

// coarse_w (NQ, CBV, DIM) -> bf16 transposed+padded (NQ, DIM, CBP), zero pad
__global__ __launch_bounds__(256) void k_cvt_coarseT(
    const float* __restrict__ cw, bf16_t* __restrict__ out)
{
    long o = (long)blockIdx.x * 256 + threadIdx.x;
    long total = (long)3 * DIMM * CBP;
    if (o >= total) return;
    long q = o / ((long)DIMM * CBP);
    long rem = o - q * (long)DIMM * CBP;
    long k = rem / CBP, c = rem % CBP;
    float v = (c < CBV) ? cw[q * (long)CBV * DIMM + c * DIMM + k] : 0.f;
    out[o] = (bf16_t)v;
}

// ---------------- layernorm (row of 1024) -> bf16 ----------------
__global__ __launch_bounds__(256) void k_ln(
    const float* __restrict__ x, const float* __restrict__ g,
    const float* __restrict__ bb, bf16_t* __restrict__ out)
{
    long row = blockIdx.x;
    const float* xr = x + row * DIMM;
    float vals[4];
    float s = 0.f, s2 = 0.f;
    #pragma unroll
    for (int u = 0; u < 4; u++) {
        float v = xr[threadIdx.x + u * 256];
        vals[u] = v; s += v; s2 += v * v;
    }
    #pragma unroll
    for (int off = 16; off >= 1; off >>= 1) {
        s  += __shfl_xor(s,  off, 32);
        s2 += __shfl_xor(s2, off, 32);
    }
    __shared__ float rs[8], rs2[8];
    int wave = threadIdx.x >> 5, lane = threadIdx.x & 31;
    if (lane == 0) { rs[wave] = s; rs2[wave] = s2; }
    __syncthreads();
    float ts = 0.f, ts2 = 0.f;
    #pragma unroll
    for (int w = 0; w < 8; w++) { ts += rs[w]; ts2 += rs2[w]; }
    float mean = ts * (1.0f / DIMM);
    float var  = ts2 * (1.0f / DIMM) - mean * mean;
    float rstd = rsqrtf(var + 1e-5f);
    bf16_t* orow = out + row * DIMM;
    #pragma unroll
    for (int u = 0; u < 4; u++) {
        int d = threadIdx.x + u * 256;
        orow[d] = (bf16_t)((vals[u] - mean) * rstd * g[d] + bb[d]);
    }
}

// ---------------- GELU(gate)*a -> bf16, row stride FFIK, zero pad cols ------
__global__ __launch_bounds__(256) void k_gelugate(
    const float* __restrict__ h, bf16_t* __restrict__ g)
{
    long idx = (long)blockIdx.x * 256 + threadIdx.x;
    if (idx >= (long)ROWS * FFIK) return;
    long r = idx / FFIK, c = idx % FFIK;
    float res = 0.f;
    if (c < FFI) {
        float a    = h[r * (2 * FFI) + c];
        float gate = h[r * (2 * FFI) + FFI + c];
        res = 0.5f * gate * (1.0f + erff(gate * 0.70710678118f)) * a;
    }
    g[idx] = (bf16_t)res;
}

// ---------------- bf16 WMMA GEMM: C = A(MxK) * B(KxN) [+bias][+C] ----------
// Preconditions (guaranteed by buffer layout):
//   K % 32 == 0; lda >= K (A rows clamped to row 0 past M);
//   B has >= gridDim.x*128 zero-padded columns (ldb) and K rows.
__global__ __launch_bounds__(256) void k_gemm(
    const bf16_t* __restrict__ A, const bf16_t* __restrict__ Bw,
    float* __restrict__ C, const float* __restrict__ bias,
    int M, int N, int K, int lda, int ldb, int ldc,
    long strideA, long strideC, int addC)
{
    __shared__ __align__(16) bf16_t As[128 * 40];
    __shared__ __align__(16) bf16_t Bs[128 * 40];
    A += (long)blockIdx.z * strideA;
    C += (long)blockIdx.z * strideC;
    const int m0 = blockIdx.y * 128;
    const int n0 = blockIdx.x * 128;
    const int t = threadIdx.x;
    const int wave = t >> 5, lane = t & 31;
    const int half = lane >> 4, lm = lane & 15;
    const int wm = (wave & 1) * 64;
    const int wn = (wave >> 1) * 32;

    v8f acc[4][2];
    #pragma unroll
    for (int i = 0; i < 4; i++)
        #pragma unroll
        for (int j = 0; j < 2; j++)
            #pragma unroll
            for (int v = 0; v < 8; v++) acc[i][j][v] = 0.f;

    const int ar = t >> 1, ac = (t & 1) * 16;   // A: 128 rows x 32 cols
    const int bk = t >> 3, bn = (t & 7) * 16;   // B: 32 rows x 128 cols

    const int gm = m0 + ar;
    const bf16_t* Ap = A + (long)(gm < M ? gm : 0) * lda;  // clamped row (safe)
    const bf16_t* Bp = Bw + n0 + bn;

    for (int k0 = 0; k0 < K; k0 += 32) {
        // A tile: branchless, fully in-bounds by construction
        v8bf a0 = *(const v8bf*)(Ap + k0 + ac);
        v8bf a1 = *(const v8bf*)(Ap + k0 + ac + 8);
        *(v8bf*)(&As[ar * 40 + ac])     = a0;
        *(v8bf*)(&As[ar * 40 + ac + 8]) = a1;
        // B tile: branchless, zero-padded columns by construction
        long rowb = (long)(k0 + bk) * ldb;
        v8bf lo = *(const v8bf*)(Bp + rowb);
        v8bf hi = *(const v8bf*)(Bp + rowb + 8);
        #pragma unroll
        for (int e = 0; e < 8; e++) {
            Bs[(bn + e) * 40 + bk]     = lo[e];
            Bs[(bn + 8 + e) * 40 + bk] = hi[e];
        }
        if (k0 + 32 < K) {
            __builtin_prefetch(Ap + k0 + 32 + ac, 0, 3);
            __builtin_prefetch(Bp + (long)(k0 + 32 + bk) * ldb, 0, 3);
        }
        __syncthreads();

        v16bf af[4], bfm[2];
        #pragma unroll
        for (int i = 0; i < 4; i++) {   // A layout: k = half*8 + {0..7, 16..23}
            const bf16_t* base = &As[(wm + 16 * i + lm) * 40 + half * 8];
            #pragma unroll
            for (int e = 0; e < 8; e++) { af[i][e] = base[e]; af[i][8 + e] = base[16 + e]; }
        }
        #pragma unroll
        for (int j = 0; j < 2; j++) {   // B layout: k = half*8 + {0..7, 16..23}
            const bf16_t* base = &Bs[(wn + 16 * j + lm) * 40 + half * 8];
            #pragma unroll
            for (int e = 0; e < 16; e++) bfm[j][e] = base[e < 8 ? e : e + 8];
        }
        #pragma unroll
        for (int i = 0; i < 4; i++)
            #pragma unroll
            for (int j = 0; j < 2; j++)
                acc[i][j] = wmma_bf16(af[i], bfm[j], acc[i][j]);
        __syncthreads();
    }

    const bool fullTile = (m0 + 127 < M) && (n0 + 127 < N);
    if (fullTile) {
        #pragma unroll
        for (int i = 0; i < 4; i++) {
            #pragma unroll
            for (int j = 0; j < 2; j++) {
                int col = n0 + wn + 16 * j + lm;
                float bv = bias ? bias[col] : 0.f;
                #pragma unroll
                for (int v = 0; v < 8; v++) {
                    int row = m0 + wm + 16 * i + v + half * 8;
                    long idx = (long)row * ldc + col;
                    float r = acc[i][j][v] + bv;
                    if (addC) r += C[idx];
                    C[idx] = r;
                }
            }
        }
    } else {
        #pragma unroll
        for (int i = 0; i < 4; i++) {
            #pragma unroll
            for (int j = 0; j < 2; j++) {
                int col = n0 + wn + 16 * j + lm;
                #pragma unroll
                for (int v = 0; v < 8; v++) {
                    int row = m0 + wm + 16 * i + v + half * 8;
                    if (row < M && col < N) {
                        float r = acc[i][j][v];
                        if (bias) r += bias[col];
                        long idx = (long)row * ldc + col;
                        if (addC) r += C[idx];
                        C[idx] = r;
                    }
                }
            }
        }
    }
}

// ---------------- flash attention: one wave per 16-query tile ----------------
__global__ __launch_bounds__(32) void k_attn(
    const float* __restrict__ qbuf, const float* __restrict__ kvbuf,
    const float* __restrict__ btab, bf16_t* __restrict__ obuf)
{
    const int i0 = blockIdx.x * 16;
    const int h  = blockIdx.y;
    const int b  = blockIdx.z;
    const int lane = threadIdx.x;
    const int half = lane >> 4, lm = lane & 15;

    __shared__ float bt[NTOK];
    for (int d = lane; d < NTOK; d += 32) bt[d] = btab[h * NTOK + d];
    __shared__ __align__(16) bf16_t Pl[16 * 32];
    __syncthreads();

    const float* Q  = qbuf + (long)b * NTOK * 512 + (long)h * 64;
    const float* KV = kvbuf + (long)b * NTOK * 128;

    int qrow = i0 + lm; if (qrow > NTOK - 1) qrow = NTOK - 1;
    v16bf qf[2];
    #pragma unroll
    for (int f = 0; f < 2; f++) {       // A layout k = half*8 + {0..7,16..23}
        const float* src = Q + (long)qrow * 512 + f * 32 + half * 8;
        #pragma unroll
        for (int e = 0; e < 8; e++) { qf[f][e] = (bf16_t)src[e]; qf[f][8 + e] = (bf16_t)src[16 + e]; }
    }

    float mrow[8], lrow[8];
    v8f ot[4];
    #pragma unroll
    for (int v = 0; v < 8; v++) {
        mrow[v] = -3.0e38f; lrow[v] = 0.f;
        ot[0][v] = 0.f; ot[1][v] = 0.f; ot[2][v] = 0.f; ot[3][v] = 0.f;
    }
    const float sc = 0.125f;            // DIM_HEAD^-0.5
    int jmax = i0 + 16; if (jmax > NTOK) jmax = NTOK;

    for (int j0 = 0; j0 < jmax; j0 += 32) {
        v8f S[2];
        #pragma unroll
        for (int jt = 0; jt < 2; jt++) {
            #pragma unroll
            for (int v = 0; v < 8; v++) S[jt][v] = 0.f;
            int jc = j0 + jt * 16 + lm; if (jc > NTOK - 1) jc = NTOK - 1;
            #pragma unroll
            for (int kk = 0; kk < 2; kk++) {    // B layout: k = kk*32 + half*8 + {0..7,16..23}
                const float* src = KV + (long)jc * 128 + kk * 32 + half * 8;
                v16bf kf;
                #pragma unroll
                for (int e = 0; e < 16; e++) kf[e] = (bf16_t)src[e < 8 ? e : e + 8];
                S[jt] = wmma_bf16(qf[kk], kf, S[jt]);
            }
        }
        // scale + bias + causal mask, online softmax update
        float sval[2][8], rmax[8], scl[8];
        #pragma unroll
        for (int v = 0; v < 8; v++) {
            int i = i0 + v + half * 8;
            float mx = -3.0e38f;
            #pragma unroll
            for (int jt = 0; jt < 2; jt++) {
                int j = j0 + jt * 16 + lm;
                float s = S[jt][v] * sc;
                int dd = i - j; if (dd < 0) dd = 0;
                s += bt[dd];
                if (j > i || j >= NTOK || i >= NTOK) s = -3.0e38f;
                sval[jt][v] = s;
                mx = fmaxf(mx, s);
            }
            #pragma unroll
            for (int off = 8; off >= 1; off >>= 1) mx = fmaxf(mx, __shfl_xor(mx, off, 16));
            rmax[v] = mx;
        }
        #pragma unroll
        for (int v = 0; v < 8; v++) {
            float mn = fmaxf(mrow[v], rmax[v]);
            scl[v] = __expf(mrow[v] - mn);
            mrow[v] = mn;
            float rsum = 0.f;
            #pragma unroll
            for (int jt = 0; jt < 2; jt++) {
                float pv = __expf(sval[jt][v] - mn);
                sval[jt][v] = pv;
                rsum += pv;
            }
            #pragma unroll
            for (int off = 8; off >= 1; off >>= 1) rsum += __shfl_xor(rsum, off, 16);
            lrow[v] = lrow[v] * scl[v] + rsum;
            ot[0][v] *= scl[v]; ot[1][v] *= scl[v]; ot[2][v] *= scl[v]; ot[3][v] *= scl[v];
        }
        // P (C-layout) -> LDS -> A-layout fragment
        __syncthreads();
        #pragma unroll
        for (int v = 0; v < 8; v++)
            #pragma unroll
            for (int jt = 0; jt < 2; jt++)
                Pl[(v + half * 8) * 32 + jt * 16 + lm] = (bf16_t)sval[jt][v];
        __syncthreads();
        v16bf pf;
        {
            const bf16_t* base = &Pl[lm * 32 + half * 8];
            #pragma unroll
            for (int e = 0; e < 8; e++) { pf[e] = base[e]; pf[8 + e] = base[16 + e]; }
        }
        #pragma unroll
        for (int dt = 0; dt < 4; dt++) {
            v16bf vf;   // B layout: k(=j) = half*8 + {0..7,16..23}, n(=d) = dt*16 + lm
            #pragma unroll
            for (int e = 0; e < 16; e++) {
                int j = j0 + half * 8 + (e < 8 ? e : e + 8); if (j > NTOK - 1) j = NTOK - 1;
                vf[e] = (bf16_t)KV[(long)j * 128 + 64 + dt * 16 + lm];
            }
            ot[dt] = wmma_bf16(pf, vf, ot[dt]);
        }
    }
    #pragma unroll
    for (int v = 0; v < 8; v++) {
        int i = i0 + v + half * 8;
        if (i < NTOK) {
            float inv = 1.0f / lrow[v];
            #pragma unroll
            for (int dt = 0; dt < 4; dt++)
                obuf[((long)b * NTOK + i) * 512 + h * 64 + dt * 16 + lm] =
                    (bf16_t)(ot[dt][v] * inv);
        }
    }
}

// =====================================================================
extern "C" void kernel_launch(void* const* d_in, const int* in_sizes, int n_in,
                              void* d_out, int out_size, void* d_ws, size_t ws_size,
                              hipStream_t stream) {
    const int*   sem_ids    = (const int*)d_in[0];
    const int*   coarse_ids = (const int*)d_in[1];
    const float* start_tok  = (const float*)d_in[2];
    const float* sem_emb    = (const float*)d_in[3];
    const float* coarse_emb = (const float*)d_in[4];
    const float* rel_emb    = (const float*)d_in[5];
    const float* attn_g     = (const float*)d_in[6];
    const float* attn_b     = (const float*)d_in[7];
    const float* wq         = (const float*)d_in[8];
    const float* wkv        = (const float*)d_in[9];
    const float* wo         = (const float*)d_in[10];
    const float* ff_g       = (const float*)d_in[11];
    const float* ff_b       = (const float*)d_in[12];
    const float* w1         = (const float*)d_in[13];
    const float* w2         = (const float*)d_in[14];
    const float* final_g    = (const float*)d_in[15];
    const float* final_b    = (const float*)d_in[16];
    const float* sem_w      = (const float*)d_in[17];
    const float* sem_b      = (const float*)d_in[18];
    const float* coarse_w   = (const float*)d_in[19];
    float* outf = (float*)d_out;

    char* p = (char*)d_ws;
    auto alloc = [&](size_t bytes) -> char* {
        char* r = p; p += (bytes + 255) & ~(size_t)255; return r;
    };
    float*  x    = (float*) alloc((size_t)ROWS * DIMM * 4);
    bf16_t* xn   = (bf16_t*)alloc((size_t)ROWS * DIMM * 2);
    float*  qb   = (float*) alloc((size_t)ROWS * 512 * 4);
    float*  kvb  = (float*) alloc((size_t)ROWS * 128 * 4);
    bf16_t* ob   = (bf16_t*)alloc((size_t)ROWS * 512 * 2);
    float*  hb   = (float*) alloc((size_t)ROWS * W1N * 4);
    bf16_t* gb   = (bf16_t*)alloc((size_t)ROWS * FFIK * 2);
    bf16_t* xf   = (bf16_t*)alloc((size_t)ROWS * DIMM * 2);
    bf16_t* swb  = (bf16_t*)alloc((size_t)DIMM * SEMP * 2);
    bf16_t* cwT  = (bf16_t*)alloc((size_t)3 * DIMM * CBP * 2);
    float*  btab = (float*) alloc((size_t)8 * NTOK * 4);
    bf16_t* wqb  = (bf16_t*)alloc((size_t)DIMM * 512 * 2);
    bf16_t* wkvb = (bf16_t*)alloc((size_t)DIMM * 128 * 2);
    bf16_t* wob  = (bf16_t*)alloc((size_t)512 * DIMM * 2);
    bf16_t* w1b  = (bf16_t*)alloc((size_t)DIMM * W1NP * 2);
    bf16_t* w2b  = (bf16_t*)alloc((size_t)FFIK * DIMM * 2);

    auto cvt = [&](const float* src, bf16_t* dst, long n) {
        k_cvt<<<dim3((unsigned)((n + 1023) / 1024)), 256, 0, stream>>>(src, dst, n);
    };
    auto cvt_pad = [&](const float* src, bf16_t* dst, int Ks, int Ns, int Kp, int Np) {
        long total = (long)Kp * Np;
        k_cvt_pad2d<<<dim3((unsigned)((total + 255) / 256)), 256, 0, stream>>>(
            src, dst, Ks, Ns, Kp, Np);
    };
    auto gemm = [&](const bf16_t* A, const bf16_t* Bw, float* C, const float* bias,
                    int M, int N, int K, int lda, int ldb, int ldc,
                    long sA, long sC, int addC, int batch) {
        dim3 g((N + 127) / 128, (M + 127) / 128, batch);
        k_gemm<<<g, 256, 0, stream>>>(A, Bw, C, bias, M, N, K, lda, ldb, ldc, sA, sC, addC);
    };

    // embeddings + bias table + constant weight conversions
    k_embed<<<dim3(ROWS), 256, 0, stream>>>(sem_ids, coarse_ids, start_tok,
                                            sem_emb, coarse_emb, x);
    k_biastab<<<dim3((8 * NTOK + 255) / 256), 256, 0, stream>>>(rel_emb, btab);
    cvt_pad(sem_w, swb, DIMM, SEMV, DIMM, SEMP);
    k_cvt_coarseT<<<dim3((unsigned)(((long)3 * DIMM * CBP + 255) / 256)), 256, 0, stream>>>(
        coarse_w, cwT);

    for (int l = 0; l < 6; l++) {
        cvt(wq  + (long)l * DIMM * 512,  wqb,  (long)DIMM * 512);
        cvt(wkv + (long)l * DIMM * 128,  wkvb, (long)DIMM * 128);
        cvt(wo  + (long)l * 512 * DIMM,  wob,  (long)512 * DIMM);
        cvt_pad(w1 + (long)l * DIMM * W1N, w1b, DIMM, W1N, DIMM, W1NP);
        cvt_pad(w2 + (long)l * FFI * DIMM, w2b, FFI, DIMM, FFIK, DIMM);

        // ---- attention ----
        k_ln<<<dim3(ROWS), 256, 0, stream>>>(x, attn_g + l * DIMM, attn_b + l * DIMM, xn);
        gemm(xn, wqb,  qb,  nullptr, ROWS, 512, DIMM, DIMM, 512, 512, 0, 0, 0, 1);
        gemm(xn, wkvb, kvb, nullptr, ROWS, 128, DIMM, DIMM, 128, 128, 0, 0, 0, 1);
        k_attn<<<dim3((NTOK + 15) / 16, 8, 2), 32, 0, stream>>>(qb, kvb, btab, ob);
        gemm(ob, wob, x, nullptr, ROWS, DIMM, 512, 512, DIMM, DIMM, 0, 0, 1, 1);

        // ---- feed forward ----
        k_ln<<<dim3(ROWS), 256, 0, stream>>>(x, ff_g + l * DIMM, ff_b + l * DIMM, xn);
        gemm(xn, w1b, hb, nullptr, ROWS, W1N, DIMM, DIMM, W1NP, W1N, 0, 0, 0, 1);
        k_gelugate<<<dim3((unsigned)(((long)ROWS * FFIK + 255) / 256)), 256, 0, stream>>>(hb, gb);
        gemm(gb, w2b, x, nullptr, ROWS, DIMM, FFIK, FFIK, DIMM, DIMM, 0, 0, 1, 1);
    }

    // ---- final LN + projections ----
    k_ln<<<dim3(ROWS), 256, 0, stream>>>(x, final_g, final_b, xf);

    // sem_logits: (B,256,501)
    gemm(xf, swb, outf, sem_b, 256, SEMV, DIMM, DIMM, SEMP, SEMV,
         (long)NTOK * DIMM, (long)256 * SEMV, 0, 2);

    // coarse_logits: (B,769,1025); row n uses coarse_w[n%3], pc row n = x[256+n]
    float* cl = outf + (long)2 * 256 * SEMV;
    for (int q = 0; q < 3; q++) {
        int M = (q == 0) ? 257 : 256;
        gemm(xf + (long)(256 + q) * DIMM, cwT + (long)q * DIMM * CBP,
             cl + (long)q * CBV, nullptr,
             M, CBV, DIMM, 3 * DIMM, CBP, 3 * CBV,
             (long)NTOK * DIMM, (long)769 * CBV, 0, 2);
    }
    (void)in_sizes; (void)n_in; (void)out_size; (void)ws_size;
}
